// WindowAttention_55070070670194
// MI455X (gfx1250) — matrix operand order repaired and checked by reading, not compiled, not run on hardware
//
#include <hip/hip_runtime.h>
#include <hip/hip_bf16.h>

typedef __bf16 bf16;
typedef __attribute__((ext_vector_type(16))) __bf16 v16bf;
typedef __attribute__((ext_vector_type(8)))  float  v8f;

#define WMMA_BF16(a, b, c) \
  __builtin_amdgcn_wmma_f32_16x16x32_bf16(false, (a), false, (b), (short)0, (c), false, false)

#define SCALE_F 0.1767766952966369f  // 32^-0.5

union FragU { v16bf v; uint4 q[2]; };
union Pack4 { uint2 u; bf16 e[4]; };
union Pack8 { uint4 u; bf16 e[8]; };

__device__ __forceinline__ v8f zero8() {
  v8f z;
  #pragma unroll
  for (int e = 0; e < 8; ++e) z[e] = 0.0f;
  return z;
}

// A-matrix 16x32 bf16 fragment from LDS row-major [rows][ldk].
// lane L holds row row0+L; elements 0..7 <- K = h*8.. ; 8..15 <- K = 16+h*8..
__device__ __forceinline__ v16bf ldsA_frag(const bf16* base, int row0, int ldk, int lane) {
  const int L = lane & 15, h = lane >> 4;
  const bf16* p = base + (row0 + L) * ldk + h * 8;
  FragU u;
  u.q[0] = *(const uint4*)(p);
  u.q[1] = *(const uint4*)(p + 16);
  return u.v;
}

// B-matrix 32x16 bf16 fragment where B(kk,n) = base[(n0+n)*ldk + kk]
// (stored transposed, K contiguous). lane n; elements 0..7 <- K = h*8.. ; 8..15 <- K = 16+h*8..
__device__ __forceinline__ v16bf ldsBT_frag(const bf16* base, int n0, int ldk, int lane) {
  const int n = lane & 15, h = lane >> 4;
  const bf16* p = base + (n0 + n) * ldk + h * 8;
  FragU u;
  u.q[0] = *(const uint4*)(p);
  u.q[1] = *(const uint4*)(p + 16);
  return u.v;
}

// gfx1250 async global->LDS copy (ASYNCcnt path)
__device__ __forceinline__ void async_ld_b128(unsigned int lds_off, const void* gptr) {
  asm volatile("global_load_async_to_lds_b128 %0, %1, off"
               :: "v"(lds_off), "v"(gptr) : "memory");
}
__device__ __forceinline__ void wait_async0() {
  asm volatile("s_wait_asynccnt 0" ::: "memory");
}

// ---------------------------------------------------------------------------
// fused per-contrast bias: cb*[m][n] = bias[n] (+ emb[sel[m]][n])
// ---------------------------------------------------------------------------
__global__ __launch_bounds__(256) void fuse_bias_kernel(
    const float* __restrict__ bq, const float* __restrict__ bk,
    const float* __restrict__ bv, const float* __restrict__ emb,
    const int* __restrict__ sq, const int* __restrict__ sk,
    float* __restrict__ cbq, float* __restrict__ cbk, float* __restrict__ cbv) {
  const int i = blockIdx.x * 256 + threadIdx.x;
  if (i >= 4 * 384) return;
  const int m = i / 384, n = i - m * 384;
  cbq[i] = bq[n] + emb[(size_t)sq[m] * 384 + n];
  cbk[i] = bk[n] + emb[(size_t)(16 + sk[m]) * 384 + n];
  cbv[i] = bv[n];
}

// ---------------------------------------------------------------------------
// Projection GEMM: Y[bn,head,i,hd] = bf16( X @ W^T + cb[m] )
// 256 thr (8 waves); tile 128x128; K loop 12 x 32; register-prefetch pipeline.
// ---------------------------------------------------------------------------
__global__ __launch_bounds__(256) void proj_kernel(
    const float* __restrict__ X, const float* __restrict__ Wm,
    const float* __restrict__ CB, bf16* __restrict__ Y) {
  __shared__ bf16 Als[128 * 32];
  __shared__ bf16 Wls[128 * 32];  // Wls[n][kk] = W[n0+n][k0+kk]
  const int tid = threadIdx.x, lane = tid & 31, wave = tid >> 5;
  const int rb = blockIdx.x, n0 = blockIdx.y * 128;

  // hoist all staging address math out of the K loop
  size_t gA[4]; int lA[4]; size_t gB[4];
  #pragma unroll
  for (int it = 0; it < 4; ++it) {
    const int i = it * 256 + tid;
    const int ar = i >> 3, c4 = (i & 7) * 4;
    const int R = rb * 128 + ar;
    const int bn = R >> 8, ii = R & 255;
    const int m = ii >> 6, sp = ii & 63;
    const int b = bn >> 6, wy = (bn >> 3) & 7, wx = bn & 7;
    const int y = wy * 8 + (sp >> 3), x = wx * 8 + (sp & 7);
    gA[it] = ((size_t)((b * 4 + m) * 64 + y) * 64 + x) * 384 + c4;
    lA[it] = ar * 32 + c4;
    gB[it] = (size_t)(n0 + ar) * 384 + c4;
  }

  v8f acc[8];
  #pragma unroll
  for (int j = 0; j < 8; ++j) acc[j] = zero8();

  // prefetch tile 0 into registers
  float4 pa4[4], pw4[4];
  #pragma unroll
  for (int it = 0; it < 4; ++it) {
    pa4[it] = *(const float4*)(X + gA[it]);
    pw4[it] = *(const float4*)(Wm + gB[it]);
  }

  for (int ks = 0; ks < 12; ++ks) {
    // commit staged tile ks to LDS
    #pragma unroll
    for (int it = 0; it < 4; ++it) {
      Pack4 a; a.e[0] = (bf16)pa4[it].x; a.e[1] = (bf16)pa4[it].y;
      a.e[2] = (bf16)pa4[it].z; a.e[3] = (bf16)pa4[it].w;
      *(uint2*)(Als + lA[it]) = a.u;
      Pack4 w; w.e[0] = (bf16)pw4[it].x; w.e[1] = (bf16)pw4[it].y;
      w.e[2] = (bf16)pw4[it].z; w.e[3] = (bf16)pw4[it].w;
      *(uint2*)(Wls + lA[it]) = w.u;
    }
    __syncthreads();
    // issue global loads for tile ks+1 while WMMAs of tile ks execute
    if (ks < 11) {
      const int k1 = (ks + 1) * 32;
      #pragma unroll
      for (int it = 0; it < 4; ++it) {
        pa4[it] = *(const float4*)(X + gA[it] + k1);
        pw4[it] = *(const float4*)(Wm + gB[it] + k1);
      }
    }
    const v16bf af = ldsA_frag(Als, wave * 16, 32, lane);
    v16bf bfr[8];
    #pragma unroll
    for (int j = 0; j < 8; ++j) bfr[j] = ldsBT_frag(Wls, j * 16, 32, lane);
    #pragma unroll
    for (int j = 0; j < 8; ++j) acc[j] = WMMA_BF16(af, bfr[j], acc[j]);
    __syncthreads();
  }

  // epilogue: bnw and m are constant within this wave's strip
  const int n15 = lane & 15, hh = lane >> 4;
  const int bnw = rb >> 1;
  const int iibase = (rb & 1) * 128 + wave * 16 + 8 * hh;
  const int m = (iibase >> 6) & 3;
  #pragma unroll
  for (int j = 0; j < 8; ++j) {
    const int n = n0 + j * 16 + n15;
    const int head = n >> 5, hd = n & 31;
    const float cbm = CB[m * 384 + n];
    const size_t y0 = (((size_t)bnw * 12 + head) * 256 + iibase) * 32 + hd;
    #pragma unroll
    for (int r = 0; r < 8; ++r)
      Y[y0 + (size_t)r * 32] = (bf16)(acc[j][r] + cbm);
  }
}

// ---------------------------------------------------------------------------
// Relative-position bias gather: out[h][qi][ki]
// ---------------------------------------------------------------------------
__global__ __launch_bounds__(256) void bias_kernel(
    const float* __restrict__ bt, const int* __restrict__ sq,
    const int* __restrict__ sk, float* __restrict__ out) {
  const int id = blockIdx.x * 256 + threadIdx.x;
  if (id >= 12 * 256 * 256) return;
  const int h = id >> 16, qi = (id >> 8) & 255, ki = id & 255;
  const int mq = qi >> 6, iq = qi & 63, nk = ki >> 6, ik = ki & 63;
  const int yq = iq >> 3, xq = iq & 7, yk = ik >> 3, xk = ik & 7;
  const int idx = (sq[mq] - sk[nk] + 3) * 225 + (yq - yk + 7) * 15 + (xq - xk + 7);
  out[id] = bt[idx * 12 + h];
}

// ---------------------------------------------------------------------------
// Attention: 1 WG = 1 (window, head); 16 waves, wave owns 16 q-rows.
// Q/K staged via global_load_async_to_lds_b128; V staged transposed [hd][k].
// ---------------------------------------------------------------------------
__global__ __launch_bounds__(512) void attn_kernel(
    const bf16* __restrict__ Qh, const bf16* __restrict__ Kh,
    const bf16* __restrict__ Vh, const float* __restrict__ Bm,
    bf16* __restrict__ Xo) {
  __shared__ bf16 qls[256 * 32];
  __shared__ bf16 kls[256 * 32];
  __shared__ bf16 vt[32 * 256];  // vt[hd][k]
  __shared__ bf16 pls[16][16 * 32];
  const int tid = threadIdx.x, lane = tid & 31, wave = tid >> 5;
  const int bn = blockIdx.x / 12, head = blockIdx.x % 12;
  const size_t seg = ((size_t)bn * 12 + head) * (256 * 32);

  {  // async copy Q, K straight into LDS (ASYNCcnt path)
    const unsigned int qoff = (unsigned int)(uintptr_t)qls;
    const unsigned int koff = (unsigned int)(uintptr_t)kls;
    const uint4* sq = (const uint4*)(Qh + seg);
    const uint4* sk = (const uint4*)(Kh + seg);
    #pragma unroll
    for (int t = 0; t < 2; ++t) {
      const int i = t * 512 + tid;
      async_ld_b128(qoff + i * 16, (const void*)(sq + i));
      async_ld_b128(koff + i * 16, (const void*)(sk + i));
    }
    // V: load rows, store transposed
    #pragma unroll
    for (int t = 0; t < 2; ++t) {
      const int i = t * 512 + tid;
      const int row = i >> 2, c8 = (i & 3) * 8;
      Pack8 p; p.u = *(const uint4*)(Vh + seg + row * 32 + c8);
      #pragma unroll
      for (int j = 0; j < 8; ++j) vt[(c8 + j) * 256 + row] = p.e[j];
    }
    wait_async0();
  }
  __syncthreads();

  // QK^T : wave's 16x256 strip as 16 C tiles; B-frags preloaded in groups of 4
  v8f c[16];
  #pragma unroll
  for (int t = 0; t < 16; ++t) c[t] = zero8();
  const v16bf aq = ldsA_frag(qls, wave * 16, 32, lane);
  #pragma unroll
  for (int g = 0; g < 4; ++g) {
    v16bf bk[4];
    #pragma unroll
    for (int t4 = 0; t4 < 4; ++t4)
      bk[t4] = ldsBT_frag(kls, (g * 4 + t4) * 16, 32, lane);
    #pragma unroll
    for (int t4 = 0; t4 < 4; ++t4)
      c[g * 4 + t4] = WMMA_BF16(aq, bk[t4], c[g * 4 + t4]);
  }

  // scale + bias (addresses linear in r and t)
  const int n15 = lane & 15, hh = lane >> 4;
  const size_t bb = ((size_t)head * 256 + wave * 16 + 8 * hh) * 256 + n15;
  #pragma unroll
  for (int t = 0; t < 16; ++t) {
    #pragma unroll
    for (int r = 0; r < 8; ++r)
      c[t][r] = c[t][r] * SCALE_F + Bm[bb + (size_t)r * 256 + t * 16];
  }

  // row softmax: local max/sum over 16 tiles, then shfl_xor across the 16-lane half
  float inv_sum[8];
  #pragma unroll
  for (int r = 0; r < 8; ++r) {
    float mx = -3.0e38f;
    #pragma unroll
    for (int t = 0; t < 16; ++t) mx = fmaxf(mx, c[t][r]);
    #pragma unroll
    for (int d = 1; d < 16; d <<= 1) mx = fmaxf(mx, __shfl_xor(mx, d, 32));
    float s = 0.0f;
    #pragma unroll
    for (int t = 0; t < 16; ++t) {
      const float e = __expf(c[t][r] - mx);
      c[t][r] = e;
      s += e;
    }
    #pragma unroll
    for (int d = 1; d < 16; d <<= 1) s += __shfl_xor(s, d, 32);
    inv_sum[r] = 1.0f / s;
  }

  // PV: per-wave LDS chunk transposes P (C-layout -> A-layout), V from vt
  v8f o0 = zero8(), o1 = zero8();
  bf16* pw = &pls[wave][0];
  #pragma unroll
  for (int kk = 0; kk < 8; ++kk) {
    #pragma unroll
    for (int tt = 0; tt < 2; ++tt) {
      const int t = kk * 2 + tt;
      #pragma unroll
      for (int r = 0; r < 8; ++r)
        pw[(r + 8 * hh) * 32 + tt * 16 + n15] = (bf16)(c[t][r] * inv_sum[r]);
    }
    const v16bf ap = ldsA_frag(pw, 0, 32, lane);
    const v16bf b0 = ldsBT_frag(vt + kk * 32, 0, 256, lane);   // B(kk2,n)=vt[n][kk*32+kk2]
    const v16bf b1 = ldsBT_frag(vt + kk * 32, 16, 256, lane);
    o0 = WMMA_BF16(ap, b0, o0);
    o1 = WMMA_BF16(ap, b1, o1);
  }

  // window-reverse scatter: m, y constant per wave/half; x advances with r
  const int b = bn >> 6, wy = (bn >> 3) & 7, wx = bn & 7;
  const int m = wave >> 2;
  const int y = wy * 8 + (wave & 3) * 2 + hh;
  const size_t base0 =
      ((size_t)((b * 4 + m) * 64 + y) * 64 + wx * 8) * 384 + head * 32 + n15;
  #pragma unroll
  for (int r = 0; r < 8; ++r) {
    Xo[base0 + (size_t)r * 384] = (bf16)o0[r];
    Xo[base0 + (size_t)r * 384 + 16] = (bf16)o1[r];
  }
}

// ---------------------------------------------------------------------------
// Output GEMM: d_out = X(bf16) @ Wo^T + bo, rows linear in global layout.
// ---------------------------------------------------------------------------
__global__ __launch_bounds__(256) void out_gemm(
    const bf16* __restrict__ X, const float* __restrict__ Wm,
    const float* __restrict__ bias, float* __restrict__ Yout) {
  __shared__ bf16 Als[128 * 32];
  __shared__ bf16 Wls[128 * 32];
  const int tid = threadIdx.x, lane = tid & 31, wave = tid >> 5;
  const int rb = blockIdx.x, n0 = blockIdx.y * 128;

  size_t gA[2]; int lA[2]; size_t gB[4]; int lB[4];
  #pragma unroll
  for (int it = 0; it < 2; ++it) {
    const int i = it * 256 + tid;
    const int ar = i >> 2, c8 = (i & 3) * 8;
    gA[it] = (size_t)(rb * 128 + ar) * 384 + c8;
    lA[it] = ar * 32 + c8;
  }
  #pragma unroll
  for (int it = 0; it < 4; ++it) {
    const int i = it * 256 + tid;
    const int nl = i >> 3, kq = (i & 7) * 4;
    gB[it] = (size_t)(n0 + nl) * 384 + kq;
    lB[it] = nl * 32 + kq;
  }

  v8f acc[8];
  #pragma unroll
  for (int j = 0; j < 8; ++j) acc[j] = zero8();

  uint4 pa[2]; float4 pw4[4];
  #pragma unroll
  for (int it = 0; it < 2; ++it) pa[it] = *(const uint4*)(X + gA[it]);
  #pragma unroll
  for (int it = 0; it < 4; ++it) pw4[it] = *(const float4*)(Wm + gB[it]);

  for (int ks = 0; ks < 12; ++ks) {
    #pragma unroll
    for (int it = 0; it < 2; ++it) *(uint4*)(Als + lA[it]) = pa[it];
    #pragma unroll
    for (int it = 0; it < 4; ++it) {
      Pack4 w; w.e[0] = (bf16)pw4[it].x; w.e[1] = (bf16)pw4[it].y;
      w.e[2] = (bf16)pw4[it].z; w.e[3] = (bf16)pw4[it].w;
      *(uint2*)(Wls + lB[it]) = w.u;
    }
    __syncthreads();
    if (ks < 11) {
      const int k1 = (ks + 1) * 32;
      #pragma unroll
      for (int it = 0; it < 2; ++it) pa[it] = *(const uint4*)(X + gA[it] + k1);
      #pragma unroll
      for (int it = 0; it < 4; ++it) pw4[it] = *(const float4*)(Wm + gB[it] + k1);
    }
    const v16bf af = ldsA_frag(Als, wave * 16, 32, lane);
    v16bf bfr[8];
    #pragma unroll
    for (int j = 0; j < 8; ++j) bfr[j] = ldsBT_frag(Wls, j * 16, 32, lane);
    #pragma unroll
    for (int j = 0; j < 8; ++j) acc[j] = WMMA_BF16(af, bfr[j], acc[j]);
    __syncthreads();
  }

  const int n15 = lane & 15, hh = lane >> 4;
  const int R0 = rb * 128 + wave * 16 + 8 * hh;
  #pragma unroll
  for (int j = 0; j < 8; ++j) {
    const int n = n0 + j * 16 + n15;
    const float bv_ = bias[n];
    const size_t a0 = (size_t)R0 * 384 + n;
    #pragma unroll
    for (int r = 0; r < 8; ++r)
      Yout[a0 + (size_t)r * 384] = acc[j][r] + bv_;
  }
}

// ---------------------------------------------------------------------------
extern "C" void kernel_launch(void* const* d_in, const int* in_sizes, int n_in,
                              void* d_out, int out_size, void* d_ws, size_t ws_size,
                              hipStream_t stream) {
  (void)in_sizes; (void)n_in; (void)out_size; (void)ws_size;
  const float* q   = (const float*)d_in[0];
  const float* k   = (const float*)d_in[1];
  const float* v   = (const float*)d_in[2];
  const int* sel_q = (const int*)d_in[3];
  const int* sel_k = (const int*)d_in[4];
  const float* Wq = (const float*)d_in[5];  const float* bq = (const float*)d_in[6];
  const float* Wk = (const float*)d_in[7];  const float* bk = (const float*)d_in[8];
  const float* Wv = (const float*)d_in[9];  const float* bv = (const float*)d_in[10];
  const float* Wo = (const float*)d_in[11]; const float* bo = (const float*)d_in[12];
  const float* btab = (const float*)d_in[13];
  const float* emb  = (const float*)d_in[14];

  char* ws = (char*)d_ws;
  bf16* qh  = (bf16*)(ws + 0);           // 25,165,824 B
  bf16* kh  = (bf16*)(ws + 25165824);    // 25,165,824 B
  bf16* vh  = (bf16*)(ws + 50331648);    // 25,165,824 B
  float* bm = (float*)(ws + 75497472);   //  3,145,728 B
  bf16* xo  = (bf16*)(ws + 78643200);    // 25,165,824 B
  float* cbq = (float*)(ws + 103809024); //      6,144 B
  float* cbk = (float*)(ws + 103815168); //      6,144 B
  float* cbv = (float*)(ws + 103821312); //      6,144 B

  fuse_bias_kernel<<<6, 256, 0, stream>>>(bq, bk, bv, emb, sel_q, sel_k, cbq, cbk, cbv);
  bias_kernel<<<3072, 256, 0, stream>>>(btab, sel_q, sel_k, bm);

  dim3 g(256, 3, 1);
  proj_kernel<<<g, 256, 0, stream>>>(q, Wq, cbq, qh);
  proj_kernel<<<g, 256, 0, stream>>>(k, Wk, cbk, kh);
  proj_kernel<<<g, 256, 0, stream>>>(v, Wv, cbv, vh);
  attn_kernel<<<1536, 512, 0, stream>>>(qh, kh, vh, bm, xo);
  out_gemm<<<g, 256, 0, stream>>>(xo, Wo, bo, (float*)d_out);
}
